// DynamiSE_57183194579704
// MI455X (gfx1250) — matrix operand order repaired and checked by reading, not compiled, not run on hardware
//
#include <hip/hip_runtime.h>
#include <hip/hip_bf16.h>
#include <math.h>

// ---------------------------------------------------------------------------
// DynamiSE on MI455X (gfx1250, wave32).
//  - Dense GEMMs via v_wmma_f32_16x16x32_f16 (f16 in, f32 accumulate);
//    one wave produces a full 16x32 row block (shared A fragment, 2 WMMAs).
//  - Sparse GCN aggregation via f32 global atomics; warp-uniform edge
//    indices are loaded once and broadcast via shfl.
//  - RK4 stages fused into a single finalize kernel per conv.
// ---------------------------------------------------------------------------

typedef __attribute__((ext_vector_type(16))) _Float16 v16h;
typedef __attribute__((ext_vector_type(8)))  float    v8f;

#define HID 32

// ---- WMMA fragment helpers (CDNA5 layouts, CDNA5 ISA §7.12.2) -------------

// A fragment: 16x32 f16 tile from row-major f32 matrix X (ldx cols),
// rows row0..row0+15, columns kbase..kbase+31.
__device__ __forceinline__ v16h load_a_frag(const float* __restrict__ X, int ldx,
                                            int row0, int kbase, int lane) {
    int r  = row0 + (lane & 15);
    int k0 = kbase + ((lane >> 4) << 3);     // lanes 0-15: +0 ; lanes 16-31: +8
    const float* p = X + (size_t)r * ldx;
    v16h a;
#pragma unroll
    for (int j = 0; j < 8; ++j) {
        a[j]     = (_Float16)p[k0 + j];       // K = k0 .. k0+7
        a[8 + j] = (_Float16)p[k0 + 16 + j];  // K = k0+16 .. k0+23
    }
    return a;
}

// B fragment: 32x16 f16 tile from row-major f32 matrix W (ldw cols),
// rows krow0..krow0+31, columns col0..col0+15.
__device__ __forceinline__ v16h load_b_frag(const float* __restrict__ W, int ldw,
                                            int krow0, int col0, int lane) {
    int n  = lane & 15;
    int k0 = krow0 + ((lane >> 4) << 3);     // lanes 0-15: K 0-7,16-23 ; 16-31: K 8-15,24-31
    v16h b;
#pragma unroll
    for (int i = 0; i < 16; ++i)
        b[i] = (_Float16)W[(size_t)(k0 + i + ((i >> 3) << 3)) * ldw + col0 + n];
    return b;
}

__device__ __forceinline__ v8f wmma_f16(v16h a, v16h b, v8f c) {
    return __builtin_amdgcn_wmma_f32_16x16x32_f16(
        /*neg_a=*/false, a, /*neg_b=*/false, b,
        /*c_mod=*/(short)0, c, /*reuse_a=*/false, /*reuse_b=*/false);
}

// D store: element i -> row 8*(lane/16)+i, col lane%16.
__device__ __forceinline__ void store_d(float* __restrict__ out, int ldo,
                                        int row0, int col0, int lane, v8f c) {
    int n = lane & 15;
    int m = ((lane >> 4) << 3);
#pragma unroll
    for (int i = 0; i < 8; ++i)
        out[(size_t)(row0 + m + i) * ldo + col0 + n] = c[i];
}

// ---- GEMM kernels: one wave computes a 16x32 row block ---------------------

// out[N,32] = X[N,32] @ W[32,32]
__global__ void gemm32_kernel(const float* __restrict__ X, const float* __restrict__ W,
                              float* __restrict__ out, int nrows) {
    int wave = (int)((blockIdx.x * (size_t)blockDim.x + threadIdx.x) >> 5);
    int lane = threadIdx.x & 31;
    int ntiles = nrows >> 4;                  // row tiles
    if (wave >= ntiles) return;
    int row0 = wave * 16;
    v16h a  = load_a_frag(X, HID, row0, 0, lane);
    v16h b0 = load_b_frag(W, HID, 0, 0,  lane);
    v16h b1 = load_b_frag(W, HID, 0, 16, lane);
    v8f c0 = {}, c1 = {};
    c0 = wmma_f16(a, b0, c0);
    c1 = wmma_f16(a, b1, c1);
    store_d(out, HID, row0, 0,  lane, c0);
    store_d(out, HID, row0, 16, lane, c1);
}

// out[N,32] = X[N,128] @ W[128,32]  (K looped in 4 chunks of 32)
__global__ void gemm_init_kernel(const float* __restrict__ X, const float* __restrict__ W,
                                 float* __restrict__ out, int nrows) {
    int wave = (int)((blockIdx.x * (size_t)blockDim.x + threadIdx.x) >> 5);
    int lane = threadIdx.x & 31;
    int ntiles = nrows >> 4;
    if (wave >= ntiles) return;
    int row0 = wave * 16;
    v8f c0 = {}, c1 = {};
#pragma unroll
    for (int kk = 0; kk < 4; ++kk) {
        v16h a  = load_a_frag(X, 128, row0, kk * 32, lane);
        v16h b0 = load_b_frag(W, HID, kk * 32, 0,  lane);
        v16h b1 = load_b_frag(W, HID, kk * 32, 16, lane);
        c0 = wmma_f16(a, b0, c0);
        c1 = wmma_f16(a, b1, c1);
    }
    store_d(out, HID, row0, 0,  lane, c0);
    store_d(out, HID, row0, 16, lane, c1);
}

// Z[N,32] = concat(Zp,Zn)[N,64] @ Wc[64,32] + bias
__global__ void gemm_comb_kernel(const float* __restrict__ Zp, const float* __restrict__ Zn,
                                 const float* __restrict__ Wc, const float* __restrict__ bias,
                                 float* __restrict__ out, int nrows) {
    int wave = (int)((blockIdx.x * (size_t)blockDim.x + threadIdx.x) >> 5);
    int lane = threadIdx.x & 31;
    int ntiles = nrows >> 4;
    if (wave >= ntiles) return;
    int row0 = wave * 16;
    v8f c0 = {}, c1 = {};
    {
        v16h a  = load_a_frag(Zp, HID, row0, 0, lane);
        v16h b0 = load_b_frag(Wc, HID, 0, 0,  lane);
        v16h b1 = load_b_frag(Wc, HID, 0, 16, lane);
        c0 = wmma_f16(a, b0, c0);
        c1 = wmma_f16(a, b1, c1);
    }
    {
        v16h a  = load_a_frag(Zn, HID, row0, 0, lane);
        v16h b0 = load_b_frag(Wc, HID, 32, 0,  lane);
        v16h b1 = load_b_frag(Wc, HID, 32, 16, lane);
        c0 = wmma_f16(a, b0, c0);
        c1 = wmma_f16(a, b1, c1);
    }
    int n = lane & 15, m = ((lane >> 4) << 3);
    float ba = bias[n];
    float bb = bias[16 + n];
#pragma unroll
    for (int i = 0; i < 8; ++i) {
        out[(size_t)(row0 + m + i) * HID + n]      = c0[i] + ba;
        out[(size_t)(row0 + m + i) * HID + 16 + n] = c1[i] + bb;
    }
}

// ---- Degree / norm kernels -------------------------------------------------

__global__ void fill_kernel(float* __restrict__ p, float v, int n) {
    int i = (int)(blockIdx.x * (size_t)blockDim.x + threadIdx.x);
    if (i < n) p[i] = v;
}

__global__ void deg_kernel(const int* __restrict__ dst, float* __restrict__ deg, int ne) {
    int i = (int)(blockIdx.x * (size_t)blockDim.x + threadIdx.x);
    if (i < ne) atomicAdd(&deg[dst[i]], 1.0f);
}

__global__ void rsqrt_kernel(float* __restrict__ p, int n) {
    int i = (int)(blockIdx.x * (size_t)blockDim.x + threadIdx.x);
    if (i < n) p[i] = rsqrtf(p[i]);
}

// ---- Sparse aggregation: agg[dst] += h[src] * dinv[src]*dinv[dst] ----------
// 32 consecutive lanes = one edge (one feature per lane); warp-uniform edge
// indices/coefficient loaded by lanes 0-1 and broadcast via shfl.
__global__ void scatter_kernel(const int* __restrict__ src, const int* __restrict__ dst,
                               const float* __restrict__ h, const float* __restrict__ dinv,
                               float* __restrict__ agg, int nedges) {
    long long tid = blockIdx.x * (long long)blockDim.x + threadIdx.x;
    int e = (int)(tid >> 5);
    int f = (int)(tid & 31);
    if (e >= nedges) return;
    int idx = 0;
    if (f == 0)      idx = src[e];
    else if (f == 1) idx = dst[e];
    int s = __shfl(idx, 0, 32);
    int d = __shfl(idx, 1, 32);
    float cc = 0.0f;
    if (f == 0) cc = dinv[s] * dinv[d];
    float coef = __shfl(cc, 0, 32);
    atomicAdd(&agg[(size_t)d * HID + f], h[(size_t)s * HID + f] * coef);
}

// ---- Epilogues -------------------------------------------------------------

// init GCN: H = relu(agg + xw*dinv^2 + b)
__global__ void init_finalize_kernel(const float* __restrict__ agg, const float* __restrict__ xw,
                                     const float* __restrict__ dinv, const float* __restrict__ bias,
                                     float* __restrict__ H, int n32) {
    int i = (int)(blockIdx.x * (size_t)blockDim.x + threadIdx.x);
    if (i >= n32) return;
    int node = i >> 5, f = i & 31;
    float di = dinv[node];
    float v = agg[i] + xw[i] * di * di + bias[f];
    H[i] = fmaxf(v, 0.0f);
}

// ODE stage: k = relu(agg + h*dinv^2 + b) * sigmoid(t*wt)
//            acc = beta*acc + wacc*k ; xstage = xbase + cstage*k
__global__ void ode_finalize_kernel(const float* __restrict__ agg, const float* __restrict__ hbuf,
                                    const float* __restrict__ dinv, const float* __restrict__ bias,
                                    const float* __restrict__ wt, float t,
                                    const float* __restrict__ xbase,
                                    float* __restrict__ acc, float* __restrict__ xstage,
                                    float beta, float wacc, float cstage, int n32) {
    int i = (int)(blockIdx.x * (size_t)blockDim.x + threadIdx.x);
    if (i >= n32) return;
    int node = i >> 5, f = i & 31;
    float di = dinv[node];
    float v = agg[i] + hbuf[i] * di * di + bias[f];
    v = fmaxf(v, 0.0f);
    float gate = 1.0f / (1.0f + __expf(-t * wt[f]));
    float k = v * gate;
    acc[i] = beta * acc[i] + wacc * k;
    xstage[i] = xbase[i] + cstage * k;
}

// X += s * acc
__global__ void advance_kernel(float* __restrict__ X, const float* __restrict__ acc,
                               float s, int n32) {
    int i = (int)(blockIdx.x * (size_t)blockDim.x + threadIdx.x);
    if (i < n32) X[i] += s * acc[i];
}

// layernorm over 32 features; one wave per row
__global__ void layernorm_kernel(const float* __restrict__ Z, const float* __restrict__ g,
                                 const float* __restrict__ bln, float* __restrict__ out,
                                 int nrows) {
    int wave = (int)((blockIdx.x * (size_t)blockDim.x + threadIdx.x) >> 5);
    int lane = threadIdx.x & 31;
    if (wave >= nrows) return;
    float v = Z[(size_t)wave * HID + lane];
    float s = v;
#pragma unroll
    for (int m = 16; m >= 1; m >>= 1) s += __shfl_xor(s, m, 32);
    float mu = s * (1.0f / 32.0f);
    float d = v - mu;
    float q = d * d;
#pragma unroll
    for (int m = 16; m >= 1; m >>= 1) q += __shfl_xor(q, m, 32);
    float var = q * (1.0f / 32.0f);
    out[(size_t)wave * HID + lane] = d * rsqrtf(var + 1e-5f) * g[lane] + bln[lane];
}

// ---------------------------------------------------------------------------

static inline int cdiv(long long a, long long b) { return (int)((a + b - 1) / b); }

extern "C" void kernel_launch(void* const* d_in, const int* in_sizes, int n_in,
                              void* d_out, int out_size, void* d_ws, size_t ws_size,
                              hipStream_t stream) {
    (void)n_in; (void)out_size; (void)ws_size;

    const float* H_t    = (const float*)d_in[0];
    const int*   Apos   = (const int*)  d_in[1];
    const int*   Aneg   = (const int*)  d_in[2];
    const int*   dApos  = (const int*)  d_in[3];
    const int*   dAneg  = (const int*)  d_in[4];
    const float* W_init = (const float*)d_in[5];
    const float* b_init = (const float*)d_in[6];
    const float* W_pos  = (const float*)d_in[7];
    const float* b_pos  = (const float*)d_in[8];
    const float* wt_pos = (const float*)d_in[9];
    const float* W_neg  = (const float*)d_in[10];
    const float* b_neg  = (const float*)d_in[11];
    const float* wt_neg = (const float*)d_in[12];
    const float* W_comb = (const float*)d_in[13];
    const float* b_comb = (const float*)d_in[14];
    const float* ln_g   = (const float*)d_in[15];
    const float* ln_b   = (const float*)d_in[16];

    const int N  = in_sizes[0] / 128;    // 100000 (divisible by 16)
    const int E  = in_sizes[1] / 2;      // 1.6M
    const int DE = in_sizes[3] / 2;      // 800k
    const int N32 = N * HID;
    const size_t fbytes = (size_t)N32 * sizeof(float);

    // workspace carve-up
    float* ws = (float*)d_ws;
    float* dinv_all = ws;            ws += N;
    float* dinv_pos = ws;            ws += N;
    float* dinv_neg = ws;            ws += N;
    float* Hbuf   = ws;              ws += N32;   // init GCN output (x0)
    float* Zpos   = ws;              ws += N32;   // evolving pos state
    float* Zneg   = ws;              ws += N32;   // evolving neg state
    float* acc    = ws;              ws += N32;   // RK4 accumulator
    float* xstage = ws;              ws += N32;   // RK4 stage input
    float* hmm    = ws;              ws += N32;   // GEMM output (x @ W)
    float* agg    = ws;              ws += N32;   // scatter accumulator

    const int TB = 256;
    const int ntiles   = N >> 4;                  // 16-row blocks
    const int gGemm    = cdiv(ntiles, TB / 32);   // 8 waves per block
    const int gElem    = cdiv(N32, TB);
    const int gNode    = cdiv(N, TB);

    // ---- degrees (deg starts at 1 == self loop) ----
    fill_kernel<<<gNode, TB, 0, stream>>>(dinv_all, 1.0f, N);
    fill_kernel<<<gNode, TB, 0, stream>>>(dinv_pos, 1.0f, N);
    fill_kernel<<<gNode, TB, 0, stream>>>(dinv_neg, 1.0f, N);
    deg_kernel<<<cdiv(E, TB),  TB, 0, stream>>>(Apos + E,   dinv_all, E);
    deg_kernel<<<cdiv(E, TB),  TB, 0, stream>>>(Aneg + E,   dinv_all, E);
    deg_kernel<<<cdiv(DE, TB), TB, 0, stream>>>(dApos + DE, dinv_pos, DE);
    deg_kernel<<<cdiv(DE, TB), TB, 0, stream>>>(dAneg + DE, dinv_neg, DE);
    rsqrt_kernel<<<gNode, TB, 0, stream>>>(dinv_all, N);
    rsqrt_kernel<<<gNode, TB, 0, stream>>>(dinv_pos, N);
    rsqrt_kernel<<<gNode, TB, 0, stream>>>(dinv_neg, N);

    // ---- init GCN: H = relu(gcn(H_t, W_init)) ----
    gemm_init_kernel<<<gGemm, TB, 0, stream>>>(H_t, W_init, hmm, N);
    hipMemsetAsync(agg, 0, fbytes, stream);
    {
        long long st = (long long)E * 32;
        scatter_kernel<<<cdiv(st, TB), TB, 0, stream>>>(Apos, Apos + E, hmm, dinv_all, agg, E);
        scatter_kernel<<<cdiv(st, TB), TB, 0, stream>>>(Aneg, Aneg + E, hmm, dinv_all, agg, E);
    }
    init_finalize_kernel<<<gElem, TB, 0, stream>>>(agg, hmm, dinv_all, b_init, Hbuf, N32);

    hipMemcpyAsync(Zpos, Hbuf, fbytes, hipMemcpyDeviceToDevice, stream);
    hipMemcpyAsync(Zneg, Hbuf, fbytes, hipMemcpyDeviceToDevice, stream);

    // ---- one GCN-ODE conv: hmm = xin@W ; agg = scatter ; fused RK4 epilogue ----
    auto conv = [&](const float* xin, const float* W, const float* b, const float* wt,
                    const int* esrc, const int* edst, const float* dinv, int ne,
                    float t, float beta, float wacc, float cstage, const float* xbase) {
        gemm32_kernel<<<gGemm, TB, 0, stream>>>(xin, W, hmm, N);
        hipMemsetAsync(agg, 0, fbytes, stream);
        long long st = (long long)ne * 32;
        scatter_kernel<<<cdiv(st, TB), TB, 0, stream>>>(esrc, edst, hmm, dinv, agg, ne);
        ode_finalize_kernel<<<gElem, TB, 0, stream>>>(agg, hmm, dinv, b, wt, t,
                                                      xbase, acc, xstage,
                                                      beta, wacc, cstage, N32);
    };

    auto run_ode = [&](float* X, const float* W, const float* b, const float* wt,
                       const int* ei, int ne, const float* dinv) {
        const int* esrc = ei;
        const int* edst = ei + ne;
        const float h = 0.1f;
        for (int s = 0; s < 10; ++s) {
            float t0 = h * (float)s;
            // k1
            conv(X,      W, b, wt, esrc, edst, dinv, ne, t0,            0.0f, 1.0f, 0.5f * h, X);
            // k2
            conv(xstage, W, b, wt, esrc, edst, dinv, ne, t0 + 0.5f * h, 1.0f, 2.0f, 0.5f * h, X);
            // k3
            conv(xstage, W, b, wt, esrc, edst, dinv, ne, t0 + 0.5f * h, 1.0f, 2.0f, h,        X);
            // k4
            conv(xstage, W, b, wt, esrc, edst, dinv, ne, t0 + h,        1.0f, 1.0f, 0.0f,     X);
            advance_kernel<<<gElem, TB, 0, stream>>>(X, acc, h / 6.0f, N32);
        }
    };

    run_ode(Zpos, W_pos, b_pos, wt_pos, dApos, DE, dinv_pos);
    run_ode(Zneg, W_neg, b_neg, wt_neg, dAneg, DE, dinv_neg);

    // ---- combine + layernorm ----
    gemm_comb_kernel<<<gGemm, TB, 0, stream>>>(Zpos, Zneg, W_comb, b_comb, agg, N);
    layernorm_kernel<<<cdiv((long long)N * 32, TB), TB, 0, stream>>>(agg, ln_g, ln_b,
                                                                     (float*)d_out, N);
}
